// STGCN_Cpraio_25847113187565
// MI455X (gfx1250) — compile-verified, matrix operand order checked by reading, not run on hardware
//
#include <hip/hip_runtime.h>
#include <hip/hip_bf16.h>

typedef __attribute__((ext_vector_type(16))) __bf16 v16bf;
typedef __attribute__((ext_vector_type(8)))  float  v8f;
typedef int v4i_vec __attribute__((vector_size(16)));   // matches builtin param type

#define B_DIM   4
#define T_DIM   12
#define N_DIM   10000
#define F_DIM   16
#define H_DIM   128
#define G_DIM   512        // 4*H
#define E_DIM   160000
#define BN_DIM  (B_DIM * N_DIM)   // 40000
#define OUT_DIM 4

// ---------------------------------------------------------------------------
// WMMA helper: D = A(16x32 bf16) x B(32x16 bf16) + C(16x16 f32)
// ---------------------------------------------------------------------------
__device__ __forceinline__ v8f wmma_bf16(v16bf a, v16bf b, v8f c) {
  return __builtin_amdgcn_wmma_f32_16x16x32_bf16(false, a, false, b, (short)0, c,
                                                 false, false);
}

// Fast sigmoid/tanh: v_exp_f32 + v_rcp_f32 (avoid IEEE divide sequences)
__device__ __forceinline__ float sigm(float x) {
  return __builtin_amdgcn_rcpf(1.0f + __expf(-x));
}
__device__ __forceinline__ float tanh_fast(float x) {
  return 2.0f * __builtin_amdgcn_rcpf(1.0f + __expf(-2.0f * x)) - 1.0f;
}

// ---------------------------------------------------------------------------
// Async global->LDS staging (CDNA5 GLOBAL_LOAD_ASYNC_TO_LDS_B128, ASYNCcnt),
// with synchronous fallback if the builtin is unavailable.
// Builtin signature (from compiler diagnostic): param0 = v4i in AS1 (global),
// so cast via integers to typed AS pointers. Low 32 bits of a flat LDS address
// are the LDS byte offset (ISA aperture rules), global flat == AS1 address.
// ---------------------------------------------------------------------------
#if __has_builtin(__builtin_amdgcn_global_load_async_to_lds_b128)
#define HAVE_ASYNC_LDS 1
#endif

__device__ __forceinline__ void g2s_b128(void* ldst, const void* gsrc) {
#ifdef HAVE_ASYNC_LDS
  __builtin_amdgcn_global_load_async_to_lds_b128(
      (__attribute__((address_space(1))) v4i_vec*)(unsigned long long)(uintptr_t)gsrc,
      (__attribute__((address_space(3))) v4i_vec*)(unsigned int)(uintptr_t)ldst,
      0, 0);
#else
  *(uint4*)ldst = *(const uint4*)gsrc;
#endif
}

__device__ __forceinline__ void wait_async_all() {
#ifdef HAVE_ASYNC_LDS
#if __has_builtin(__builtin_amdgcn_s_wait_asynccnt)
  __builtin_amdgcn_s_wait_asynccnt(0);
#else
  asm volatile("s_wait_asynccnt 0x0" ::: "memory");
#endif
#endif
}

// ---------------------------------------------------------------------------
// Weight packing into CDNA5 WMMA B-operand layout (ISA 7.12.2, 16-bit B 32x16):
//  lane<16 : N=lane,    halfs j=0..15 hold K = [0..7, 16..23]
//  lane>=16: N=lane-16, halfs j=0..15 hold K = [8..15, 24..31]
// Tile = 512 bf16 (1 KB), tile index = kt*ntiles + nt.
// ---------------------------------------------------------------------------
__global__ void pack_whh_kernel(const float* __restrict__ w_hh, __bf16* __restrict__ dst) {
  int g = blockIdx.x * 256 + threadIdx.x;           // < 4*32*512 = 65536
  int tile = g >> 9, within = g & 511;
  int lane = within >> 4, j = within & 15;
  int kt = tile >> 5, nt = tile & 31;
  int hi = lane >> 4;
  int kk = (j & 7) + 8 * hi + 16 * (j >> 3);
  int k = kt * 32 + kk;
  int n = nt * 16 + (lane & 15);
  dst[g] = (__bf16)w_hh[n * H_DIM + k];             // B[k][n] = w_hh[n][k]
}

__global__ void pack_wih_kernel(const float* __restrict__ w_ih, __bf16* __restrict__ dst) {
  int g = blockIdx.x * 256 + threadIdx.x;           // < 32*512 = 16384 (K padded 16->32)
  int nt = g >> 9, within = g & 511;
  int lane = within >> 4, j = within & 15;
  int hi = lane >> 4;
  int kk = (j & 7) + 8 * hi + 16 * (j >> 3);
  int n = nt * 16 + (lane & 15);
  float v = (kk < F_DIM) ? w_ih[n * F_DIM + kk] : 0.0f;
  dst[g] = (__bf16)v;
}

// W stored row-major [128, Nout]; B[k][n] = W[k][n]
__global__ void pack_w_kernel(const float* __restrict__ W, __bf16* __restrict__ dst,
                              int ntiles, int Nout) {
  int g = blockIdx.x * 256 + threadIdx.x;           // < 4*ntiles*512
  if (g >= 4 * ntiles * 512) return;
  int tile = g >> 9, within = g & 511;
  int lane = within >> 4, j = within & 15;
  int kt = tile / ntiles, nt = tile % ntiles;
  int hi = lane >> 4;
  int kk = (j & 7) + 8 * hi + 16 * (j >> 3);
  int k = kt * 32 + kk;
  int n = nt * 16 + (lane & 15);
  dst[g] = (__bf16)W[k * Nout + n];
}

__global__ void bias_sum_kernel(const float* __restrict__ b_ih,
                                const float* __restrict__ b_hh,
                                float* __restrict__ bias) {
  int g = blockIdx.x * 256 + threadIdx.x;
  if (g < G_DIM) bias[g] = b_ih[g] + b_hh[g];
}

// ---------------------------------------------------------------------------
// LSTM: one block = 16 sequences, 8 waves, whole 512-gate row via WMMA.
// wave w owns gate tiles {w, w+8, w+16, w+24}  (matching i/f/g/o columns)
// and therefore h/c columns [16w, 16w+16).
// ---------------------------------------------------------------------------
__global__ __launch_bounds__(256) void lstm_kernel(
    const float* __restrict__ x,        // [B,T,N,F]
    const __bf16* __restrict__ whh_p,   // packed [4kt][32nt][512]
    const __bf16* __restrict__ wih_p,   // packed [32nt][512] (K padded to 32)
    const float* __restrict__ bias,     // [512] = b_ih + b_hh
    float* __restrict__ h_out)          // [BN, 128]
{
  __shared__ __attribute__((aligned(32))) __bf16 s_whh[4 * 32 * 512];  // 128 KB
  __shared__ __attribute__((aligned(32))) __bf16 s_hA[4 * 512];        // 4 KB
  __shared__ __attribute__((aligned(32))) __bf16 s_xA[512];            // 1 KB

  const int tid  = threadIdx.x;
  const int wave = tid >> 5;
  const int lane = tid & 31;
  const int l15  = lane & 15;
  const int mhi  = (lane >> 4) * 8;      // row offset from lane-half (C layout)
  const int row0 = blockIdx.x * 16;

  // Stage packed w_hh into LDS: async memory->LDS (no VGPR roundtrip)
  for (int i = tid; i < (4 * 32 * 512) / 8; i += 256)
    g2s_b128((void*)((uint4*)s_whh + i), (const void*)((const uint4*)whh_p + i));

  for (int i = tid; i < 4 * 512; i += 256) s_hA[i] = (__bf16)0.0f;   // h0 = 0
  for (int i = tid; i < 512; i += 256)     s_xA[i] = (__bf16)0.0f;   // K-pad = 0

  // Per-wave persistent x-weight B operands + biases
  v16bf wihB[4];
  float biasv[4];
#pragma unroll
  for (int q = 0; q < 4; ++q) {
    int nt = wave + 8 * q;
    wihB[q]  = *(const v16bf*)(wih_p + nt * 512 + lane * 16);
    biasv[q] = bias[nt * 16 + l15];
  }

  // Pack x for t=0 into A layout (16x32, K 16..31 stays zero)
  {
    int r = tid >> 4, f = tid & 15;
    int s = row0 + r;
    int batch = s / N_DIM, node = s % N_DIM;
    float v = x[(((size_t)batch * T_DIM + 0) * N_DIM + node) * F_DIM + f];
    s_xA[(r + 16 * (f >> 3)) * 16 + (f & 7)] = (__bf16)v;
  }

  float cst[8], hv[8];
#pragma unroll
  for (int r = 0; r < 8; ++r) { cst[r] = 0.0f; hv[r] = 0.0f; }

  const int hcol = 16 * wave + l15;     // h/c column owned by this lane
  const int kt_h = hcol >> 5;
  const int kk_h = hcol & 31;
  const int hi2  = (kk_h >> 3) & 1;
  const int j2   = (kk_h & 7) + 8 * ((kk_h >> 4) & 1);

  wait_async_all();                     // staged weights complete before barrier

  for (int t = 0; t < T_DIM; ++t) {
    __syncthreads();
    v16bf hA0 = *(const v16bf*)(s_hA + 0 * 512 + lane * 16);
    v16bf hA1 = *(const v16bf*)(s_hA + 1 * 512 + lane * 16);
    v16bf hA2 = *(const v16bf*)(s_hA + 2 * 512 + lane * 16);
    v16bf hA3 = *(const v16bf*)(s_hA + 3 * 512 + lane * 16);
    v16bf xA  = *(const v16bf*)(s_xA + lane * 16);

    v8f gates[4];
#pragma unroll
    for (int q = 0; q < 4; ++q) {
      int nt = wave + 8 * q;
      v8f acc = {};
      acc = wmma_bf16(xA,  wihB[q], acc);
      acc = wmma_bf16(hA0, *(const v16bf*)(s_whh + (0 * 32 + nt) * 512 + lane * 16), acc);
      acc = wmma_bf16(hA1, *(const v16bf*)(s_whh + (1 * 32 + nt) * 512 + lane * 16), acc);
      acc = wmma_bf16(hA2, *(const v16bf*)(s_whh + (2 * 32 + nt) * 512 + lane * 16), acc);
      acc = wmma_bf16(hA3, *(const v16bf*)(s_whh + (3 * 32 + nt) * 512 + lane * 16), acc);
      gates[q] = acc;
    }

#pragma unroll
    for (int r = 0; r < 8; ++r) {
      float ig = sigm(gates[0][r] + biasv[0]);
      float fg = sigm(gates[1][r] + biasv[1]);
      float gg = tanh_fast(gates[2][r] + biasv[2]);
      float og = sigm(gates[3][r] + biasv[3]);
      float c  = fg * cst[r] + ig * gg;
      cst[r] = c;
      hv[r]  = og * tanh_fast(c);
    }

    __syncthreads();  // everyone done reading s_hA of this step
#pragma unroll
    for (int r = 0; r < 8; ++r) {
      int m = r + mhi;
      s_hA[kt_h * 512 + (m + 16 * hi2) * 16 + j2] = (__bf16)hv[r];
    }
    if (t + 1 < T_DIM) {
      int r = tid >> 4, f = tid & 15;
      int s = row0 + r;
      int batch = s / N_DIM, node = s % N_DIM;
      float v = x[(((size_t)batch * T_DIM + (t + 1)) * N_DIM + node) * F_DIM + f];
      s_xA[(r + 16 * (f >> 3)) * 16 + (f & 7)] = (__bf16)v;
    }
  }

#pragma unroll
  for (int r = 0; r < 8; ++r)
    h_out[(size_t)(row0 + r + mhi) * H_DIM + hcol] = hv[r];
}

// ---------------------------------------------------------------------------
// Generic 16-row-tile GEMM:  C[M,Nout] = A[M,128] @ Bp  (+bias, optional ReLU)
// ---------------------------------------------------------------------------
__global__ __launch_bounds__(256) void gemm16_kernel(
    const float* __restrict__ A, const __bf16* __restrict__ Bp,
    const float* __restrict__ bias, float* __restrict__ C,
    int Nout, int ntiles, int relu)
{
  __shared__ __attribute__((aligned(32))) __bf16 s_A[4 * 512];
  __shared__ __attribute__((aligned(32))) __bf16 s_B[4 * 8 * 512];
  const int tid = threadIdx.x, wave = tid >> 5, lane = tid & 31, l15 = lane & 15;
  const int row0 = blockIdx.x * 16;

  // async-stage packed B tiles into LDS
  for (int i = tid; i < ntiles * 256; i += 256)
    g2s_b128((void*)((uint4*)s_B + i), (const void*)((const uint4*)Bp + i));

  for (int e = tid; e < 16 * H_DIM; e += 256) {     // pack A (f32 -> bf16, A layout)
    int r = e >> 7, k = e & 127;
    float v = A[(size_t)(row0 + r) * H_DIM + k];
    int kt = k >> 5, kk = k & 31;
    int hi = (kk >> 3) & 1, j = (kk & 7) + 8 * ((kk >> 4) & 1);
    s_A[kt * 512 + (r + 16 * hi) * 16 + j] = (__bf16)v;
  }
  wait_async_all();
  __syncthreads();

  if (wave < ntiles) {
    v8f acc = {};
#pragma unroll
    for (int kt = 0; kt < 4; ++kt) {
      v16bf a = *(const v16bf*)(s_A + kt * 512 + lane * 16);
      v16bf b = *(const v16bf*)(s_B + (kt * ntiles + wave) * 512 + lane * 16);
      acc = wmma_bf16(a, b, acc);
    }
    int n = wave * 16 + l15;
    float bv = bias ? bias[n] : 0.0f;
#pragma unroll
    for (int r = 0; r < 8; ++r) {
      int m = r + 8 * (lane >> 4);
      float v = acc[r] + bv;
      if (relu) v = fmaxf(v, 0.0f);
      C[(size_t)(row0 + m) * Nout + n] = v;
    }
  }
}

// ---------------------------------------------------------------------------
// Graph norm + message passing
// ---------------------------------------------------------------------------
__global__ void deg_init_kernel(float* __restrict__ deg) {
  int g = blockIdx.x * 256 + threadIdx.x;
  if (g < N_DIM) deg[g] = 1.0f;                     // self loop
}
__global__ void deg_edge_kernel(const int* __restrict__ edges, float* __restrict__ deg) {
  int e = blockIdx.x * 256 + threadIdx.x;
  if (e < E_DIM) atomicAdd(&deg[edges[E_DIM + e]], 1.0f);
}
__global__ void dinv_kernel(const float* __restrict__ deg, float* __restrict__ dinv) {
  int g = blockIdx.x * 256 + threadIdx.x;
  if (g < N_DIM) dinv[g] = rsqrtf(deg[g]);
}
__global__ void zero_kernel(float* __restrict__ p, long long n) {
  long long g = (long long)blockIdx.x * 256 + threadIdx.x;
  if (g < n) p[g] = 0.0f;
}

// one thread per (edge, batch, 4-float chunk): 4 global f32 atomics
__global__ void scatter_kernel(const float* __restrict__ hw, const int* __restrict__ edges,
                               const float* __restrict__ dinv, float* __restrict__ acc)
{
  long long g = (long long)blockIdx.x * 256 + threadIdx.x;  // < E*B*32
  int part = (int)(g & 31);
  long long eb = g >> 5;
  int b = (int)(eb % B_DIM);
  int e = (int)(eb / B_DIM);
  if (e >= E_DIM) return;
  int src = edges[e];
  int dst = edges[E_DIM + e];
  float norm = dinv[src] * dinv[dst];
  const float* srow = hw + ((size_t)b * N_DIM + src) * H_DIM + part * 4;
  float* drow = acc + ((size_t)b * N_DIM + dst) * H_DIM + part * 4;
  float4 v = *(const float4*)srow;
  atomicAdd(drow + 0, v.x * norm);
  atomicAdd(drow + 1, v.y * norm);
  atomicAdd(drow + 2, v.z * norm);
  atomicAdd(drow + 3, v.w * norm);
}

// out = relu(acc + hw * dinv[n]^2 + bias[c])   (self-loop folded in)
__global__ void post_kernel(const float* __restrict__ acc, const float* __restrict__ hw,
                            const float* __restrict__ dinv, const float* __restrict__ bias,
                            float* __restrict__ out)
{
  size_t g = (size_t)blockIdx.x * 256 + threadIdx.x;
  if (g >= (size_t)BN_DIM * H_DIM) return;
  int c = (int)(g & 127);
  int n = (int)((g >> 7) % N_DIM);
  float di = dinv[n];
  float v = acc[g] + hw[g] * di * di + bias[c];
  out[g] = fmaxf(v, 0.0f);
}

__global__ void fc2_kernel(const float* __restrict__ o1, const float* __restrict__ w,
                           const float* __restrict__ b, float* __restrict__ out)
{
  int g = blockIdx.x * 256 + threadIdx.x;           // < BN*4
  if (g >= BN_DIM * OUT_DIM) return;
  int s = g >> 2, c = g & 3;
  float sum = b[c];
  const float* row = o1 + (size_t)s * 64;
#pragma unroll 8
  for (int k = 0; k < 64; ++k) sum += row[k] * w[k * OUT_DIM + c];
  out[g] = sum;
}

// ---------------------------------------------------------------------------
extern "C" void kernel_launch(void* const* d_in, const int* in_sizes, int n_in,
                              void* d_out, int out_size, void* d_ws, size_t ws_size,
                              hipStream_t stream)
{
  (void)in_sizes; (void)n_in; (void)out_size; (void)ws_size;
  const float* x      = (const float*)d_in[0];
  const int*   edges  = (const int*)d_in[1];
  const float* w_ih   = (const float*)d_in[2];
  const float* w_hh   = (const float*)d_in[3];
  const float* b_ih   = (const float*)d_in[4];
  const float* b_hh   = (const float*)d_in[5];
  const float* gcn1_w = (const float*)d_in[6];
  const float* gcn1_b = (const float*)d_in[7];
  const float* gcn2_w = (const float*)d_in[8];
  const float* gcn2_b = (const float*)d_in[9];
  const float* fc1_w  = (const float*)d_in[10];
  const float* fc1_b  = (const float*)d_in[11];
  const float* fc2_w  = (const float*)d_in[12];
  const float* fc2_b  = (const float*)d_in[13];
  float* out = (float*)d_out;

  char* ws = (char*)d_ws;
  size_t off = 0;
  auto alloc = [&](size_t bytes) -> void* {
    void* p = ws + off;
    off = (off + bytes + 255) & ~(size_t)255;
    return p;
  };
  __bf16* whh_p  = (__bf16*)alloc(4 * 32 * 512 * 2);
  __bf16* wih_p  = (__bf16*)alloc(32 * 512 * 2);
  __bf16* g1_p   = (__bf16*)alloc(4 * 8 * 512 * 2);
  __bf16* g2_p   = (__bf16*)alloc(4 * 8 * 512 * 2);
  __bf16* f1_p   = (__bf16*)alloc(4 * 4 * 512 * 2);
  float*  bias   = (float*)alloc(G_DIM * 4);
  float*  deg    = (float*)alloc(N_DIM * 4);
  float*  dinv   = (float*)alloc(N_DIM * 4);
  float*  hbuf   = (float*)alloc((size_t)BN_DIM * H_DIM * 4);
  float*  hwbuf  = (float*)alloc((size_t)BN_DIM * H_DIM * 4);
  float*  accbuf = (float*)alloc((size_t)BN_DIM * H_DIM * 4);
  float*  o1buf  = (float*)alloc((size_t)BN_DIM * 64 * 4);

  const long long feat_n = (long long)BN_DIM * H_DIM;   // 5,120,000
  const int feat_blocks  = (int)((feat_n + 255) / 256); // 20000
  const int scat_blocks  = (int)(((long long)E_DIM * B_DIM * 32 + 255) / 256); // 80000

  // ---- one-time prep (per launch) ----
  pack_whh_kernel<<<256, 256, 0, stream>>>(w_hh, whh_p);
  pack_wih_kernel<<<64, 256, 0, stream>>>(w_ih, wih_p);
  pack_w_kernel<<<64, 256, 0, stream>>>(gcn1_w, g1_p, 8, 128);
  pack_w_kernel<<<64, 256, 0, stream>>>(gcn2_w, g2_p, 8, 128);
  pack_w_kernel<<<32, 256, 0, stream>>>(fc1_w, f1_p, 4, 64);
  bias_sum_kernel<<<2, 256, 0, stream>>>(b_ih, b_hh, bias);
  deg_init_kernel<<<(N_DIM + 255) / 256, 256, 0, stream>>>(deg);
  deg_edge_kernel<<<(E_DIM + 255) / 256, 256, 0, stream>>>(edges, deg);
  dinv_kernel<<<(N_DIM + 255) / 256, 256, 0, stream>>>(deg, dinv);

  // ---- LSTM over all B*N sequences ----
  lstm_kernel<<<BN_DIM / 16, 256, 0, stream>>>(x, whh_p, wih_p, bias, hbuf);

  // ---- GCN layer 1 ----
  gemm16_kernel<<<BN_DIM / 16, 256, 0, stream>>>(hbuf, g1_p, nullptr, hwbuf, 128, 8, 0);
  zero_kernel<<<feat_blocks, 256, 0, stream>>>(accbuf, feat_n);
  scatter_kernel<<<scat_blocks, 256, 0, stream>>>(hwbuf, edges, dinv, accbuf);
  post_kernel<<<feat_blocks, 256, 0, stream>>>(accbuf, hwbuf, dinv, gcn1_b, hbuf);

  // ---- GCN layer 2 ----
  gemm16_kernel<<<BN_DIM / 16, 256, 0, stream>>>(hbuf, g2_p, nullptr, hwbuf, 128, 8, 0);
  zero_kernel<<<feat_blocks, 256, 0, stream>>>(accbuf, feat_n);
  scatter_kernel<<<scat_blocks, 256, 0, stream>>>(hwbuf, edges, dinv, accbuf);
  post_kernel<<<feat_blocks, 256, 0, stream>>>(accbuf, hwbuf, dinv, gcn2_b, hbuf);

  // ---- FC head ----
  gemm16_kernel<<<BN_DIM / 16, 256, 0, stream>>>(hbuf, f1_p, fc1_b, o1buf, 64, 4, 1);
  fc2_kernel<<<(BN_DIM * OUT_DIM + 255) / 256, 256, 0, stream>>>(o1buf, fc2_w, fc2_b, out);
}
